// FastConformerAttention_54803782697280
// MI455X (gfx1250) — matrix operand order repaired and checked by reading, not compiled, not run on hardware
//
#include <hip/hip_runtime.h>
#include <hip/hip_bf16.h>

typedef __bf16 v16bf __attribute__((ext_vector_type(16)));
typedef __bf16 v8bf  __attribute__((ext_vector_type(8)));
typedef float  v8f   __attribute__((ext_vector_type(8)));

#define HID  512
#define NB   4
#define TSEQ 1024
#define NH   8
#define DK   64
#define PPOS 2047
#define PPAD 2048

__device__ __forceinline__ v16bf cat16(v8bf lo, v8bf hi) {
  return __builtin_shufflevector(lo, hi, 0,1,2,3,4,5,6,7,8,9,10,11,12,13,14,15);
}

__device__ __forceinline__ v8f wmma_bf16(v16bf a, v16bf b, v8f c) {
  return __builtin_amdgcn_wmma_f32_16x16x32_bf16(false, a, false, b, (short)0, c, false, false);
}

// ---------------- elementwise prep kernels ----------------

__global__ void cvt_bf16_kernel(const float* __restrict__ in, __bf16* __restrict__ out, int n) {
  int i = blockIdx.x * blockDim.x + threadIdx.x;
  if (i < n) out[i] = (__bf16)in[i];
}

// W (512x512 f32, row-major) -> WT (512x512 bf16) with WT[n][k] = W[k][n]
__global__ void transpose_w_kernel(const float* __restrict__ W, __bf16* __restrict__ WT) {
  int i = blockIdx.x * blockDim.x + threadIdx.x;   // over 512*512
  int r = i >> 9, c = i & (HID - 1);
  WT[(size_t)c * HID + r] = (__bf16)W[i];
}

// qf (B*T,512) f32 (bias already added) -> Qu/Qv (B,H,T,64) bf16 with pos_bias added
__global__ void scatter_q_kernel(const float* __restrict__ qf,
                                 const float* __restrict__ pbu, const float* __restrict__ pbv,
                                 __bf16* __restrict__ Qu, __bf16* __restrict__ Qv) {
  const int i  = blockIdx.x * blockDim.x + threadIdx.x;  // B*T*HID exact
  const int c  = i & (HID - 1);
  const int bt = i >> 9;
  const int t  = bt & (TSEQ - 1);
  const int b  = bt >> 10;
  const int h  = c >> 6;
  const int dd = c & (DK - 1);
  const float v = qf[i];
  const size_t o = (((size_t)(b * NH + h)) * TSEQ + t) * DK + dd;
  Qu[o] = (__bf16)(v + pbu[c]);
  Qv[o] = (__bf16)(v + pbv[c]);
}

// kf,vf (B*T,512) f32 -> Kh (B,H,T,64) bf16 ; Vt (B,H,64,T) bf16 (transposed for B-frags)
__global__ void scatter_kv_kernel(const float* __restrict__ kf, const float* __restrict__ vf,
                                  __bf16* __restrict__ Kh, __bf16* __restrict__ Vt) {
  const int i  = blockIdx.x * blockDim.x + threadIdx.x;
  const int c  = i & (HID - 1);
  const int bt = i >> 9;
  const int t  = bt & (TSEQ - 1);
  const int b  = bt >> 10;
  const int h  = c >> 6;
  const int dd = c & (DK - 1);
  Kh[(((size_t)(b * NH + h)) * TSEQ + t) * DK + dd] = (__bf16)kf[i];
  Vt[(((size_t)(b * NH + h)) * DK + dd) * TSEQ + t] = (__bf16)vf[i];
}

// pf (P,512) f32 -> Ph (H, PPAD, 64) bf16
__global__ void scatter_p_kernel(const float* __restrict__ pf, __bf16* __restrict__ Ph) {
  const int i = blockIdx.x * blockDim.x + threadIdx.x;
  if (i >= PPOS * HID) return;
  const int c  = i & (HID - 1);
  const int pi = i >> 9;
  const int h  = c >> 6;
  const int dd = c & (DK - 1);
  Ph[((size_t)h * PPAD + pi) * DK + dd] = (__bf16)pf[i];
}

// ---------------- bf16 WMMA GEMM: out(M,512) = A(M,512) @ WT^T + bias ----------------
// block = 8 waves; blockIdx.x = M/64 tile; wave w handles 64 output columns.
// Per wave: 64x64 output tile = 4x4 WMMA accumulators; per K=32 step:
// 8 A-loads + 8 B-loads (b128) feed 16 WMMAs.
__global__ __launch_bounds__(256) void gemm_bf16_kernel(
    const __bf16* __restrict__ A, const __bf16* __restrict__ WT,
    const float* __restrict__ bias, float* __restrict__ out) {
  const int lane = threadIdx.x & 31;
  const int w    = threadIdx.x >> 5;     // n-group 0..7
  const int m64  = blockIdx.x;
  const int n    = lane & 15;
  const int n0   = w * 64;
  const int kA   = (lane < 16) ? 0 : 16;
  const int kB   = (lane < 16) ? 0 : 16;

  const __bf16* abase = A  + (size_t)(m64 * 64 + n) * HID + kA;
  const __bf16* bbase = WT + (size_t)(n0 + n) * HID + kB;

  v8f acc[4][4] = {};
  for (int k = 0; k < HID; k += 32) {
    v16bf a[4], b[4];
#pragma unroll
    for (int i = 0; i < 4; ++i) {
      const __bf16* ap = abase + (size_t)(i * 16) * HID + k;
      a[i] = cat16(*(const v8bf*)ap, *(const v8bf*)(ap + 8));
    }
#pragma unroll
    for (int j = 0; j < 4; ++j) {
      const __bf16* bp = bbase + (size_t)(j * 16) * HID + k;
      b[j] = *(const v16bf*)bp;
    }
#pragma unroll
    for (int i = 0; i < 4; ++i)
#pragma unroll
      for (int j = 0; j < 4; ++j)
        acc[i][j] = wmma_bf16(a[i], b[j], acc[i][j]);
  }

  const int mo = (lane >> 4) * 8;
#pragma unroll
  for (int j = 0; j < 4; ++j) {
    const int col = n0 + j * 16 + n;
    const float bv = bias[col];
#pragma unroll
    for (int i = 0; i < 4; ++i) {
#pragma unroll
      for (int r = 0; r < 8; ++r) {
        const int row = m64 * 64 + i * 16 + r + mo;
        out[(size_t)row * HID + col] = acc[i][j][r] + bv;
      }
    }
  }
}

// ---------------- fused rel-pos flash attention ----------------
// One wave per 16 query rows of one (b,h). Streams keys in 32-wide steps.
// score[t,j] = scale*( q_u[t].k[j] + q_v[t].p[j-t+T-1] ), online softmax, O += P.V
__global__ __launch_bounds__(256) void attn_kernel(
    const __bf16* __restrict__ Qu, const __bf16* __restrict__ Qv,
    const __bf16* __restrict__ Kh, const __bf16* __restrict__ Vt,
    const __bf16* __restrict__ Ph, __bf16* __restrict__ Obuf) {
  __shared__ float  gbuf[8][16][52];   // per-wave G staging (16 x 48 used)
  __shared__ __bf16 pbuf[8][16][40];   // per-wave prob staging (16 x 32 used)

  const int lane = threadIdx.x & 31;
  const int w    = threadIdx.x >> 5;
  const int gid  = blockIdx.x * 8 + w;        // 0 .. B*H*(T/16)-1
  const int qt   = gid & (TSEQ / 16 - 1);
  const int bh   = gid >> 6;
  const int h    = bh & (NH - 1);
  const int b    = bh >> 3;
  const int t0   = qt * 16;

  const __bf16* qu = Qu + (size_t)bh * TSEQ * DK;
  const __bf16* qv = Qv + (size_t)bh * TSEQ * DK;
  const __bf16* kh = Kh + (size_t)bh * TSEQ * DK;
  const __bf16* vt = Vt + (size_t)bh * DK * TSEQ;
  const __bf16* ph = Ph + (size_t)h * PPAD * DK;

  const int n  = lane & 15;
  const int kA = (lane < 16) ? 0 : 16;
  const int kB = (lane < 16) ? 0 : 16;
  const int mo = (lane >> 4) * 8;

  // A-fragments of q_u / q_v (16 x 64, two K=32 chunks each)
  v16bf aqu[2], aqv[2];
  {
    const __bf16* qup = qu + (size_t)(t0 + n) * DK + kA;
    const __bf16* qvp = qv + (size_t)(t0 + n) * DK + kA;
#pragma unroll
    for (int c = 0; c < 2; ++c) {
      aqu[c] = cat16(*(const v8bf*)(qup + c * 32), *(const v8bf*)(qup + c * 32 + 8));
      aqv[c] = cat16(*(const v8bf*)(qvp + c * 32), *(const v8bf*)(qvp + c * 32 + 8));
    }
  }

  v8f oacc[4] = {};
  float mrow[8], lrow[8];
#pragma unroll
  for (int r = 0; r < 8; ++r) { mrow[r] = -1e30f; lrow[r] = 0.f; }
  const float scale = 0.125f;   // 1/sqrt(64)

  for (int j0 = 0; j0 < TSEQ; j0 += 32) {
    // V B-fragments depend only on j0: issue early so they overlap the
    // score WMMAs + softmax VALU work below.
    v16bf vb[4];
#pragma unroll
    for (int nt = 0; nt < 4; ++nt)
      vb[nt] = *(const v16bf*)(vt + (size_t)(nt * 16 + n) * TSEQ + j0 + kB);

    v8f s0 = {}, s1 = {};
    v8f g[3] = {};
    const int rbase = j0 - t0 + TSEQ - 16;   // >= 0, rbase+46 <= 2046
#pragma unroll
    for (int c = 0; c < 2; ++c) {
      const __bf16* k0p = kh + (size_t)(j0 + n) * DK + c * 32 + kB;
      const __bf16* k1p = kh + (size_t)(j0 + 16 + n) * DK + c * 32 + kB;
      s0 = wmma_bf16(aqu[c], *(const v16bf*)k0p, s0);
      s1 = wmma_bf16(aqu[c], *(const v16bf*)k1p, s1);
#pragma unroll
      for (int gt = 0; gt < 3; ++gt) {
        const __bf16* pp = ph + (size_t)(rbase + gt * 16 + n) * DK + c * 32 + kB;
        g[gt] = wmma_bf16(aqv[c], *(const v16bf*)pp, g[gt]);
      }
    }
    // stage G, gather rel-shift diagonals: bd[m,dj] = G[m, dj-m+15]
#pragma unroll
    for (int gt = 0; gt < 3; ++gt)
#pragma unroll
      for (int r = 0; r < 8; ++r)
        gbuf[w][r + mo][gt * 16 + n] = g[gt][r];
    asm volatile("s_wait_dscnt 0" ::: "memory");
#pragma unroll
    for (int r = 0; r < 8; ++r) {
      const int m = r + mo;
      const float bd0 = gbuf[w][m][n - m + 15];
      const float bd1 = gbuf[w][m][n - m + 31];
      s0[r] = (s0[r] + bd0) * scale;
      s1[r] = (s1[r] + bd1) * scale;
    }
    // online softmax over this 32-key step
#pragma unroll
    for (int r = 0; r < 8; ++r) {
      float mx = fmaxf(s0[r], s1[r]);
#pragma unroll
      for (int off = 1; off < 16; off <<= 1)
        mx = fmaxf(mx, __shfl_xor(mx, off, 32));
      const float mnew = fmaxf(mrow[r], mx);
      const float corr = __expf(mrow[r] - mnew);
      mrow[r] = mnew;
      const float p0 = __expf(s0[r] - mnew);
      const float p1 = __expf(s1[r] - mnew);
      s0[r] = p0; s1[r] = p1;
      float rs = p0 + p1;
#pragma unroll
      for (int off = 1; off < 16; off <<= 1)
        rs += __shfl_xor(rs, off, 32);
      lrow[r] = lrow[r] * corr + rs;
#pragma unroll
      for (int nt = 0; nt < 4; ++nt) oacc[nt][r] *= corr;
    }
    // acc layout -> A-fragment layout via LDS (bf16)
#pragma unroll
    for (int r = 0; r < 8; ++r) {
      pbuf[w][r + mo][n]      = (__bf16)s0[r];
      pbuf[w][r + mo][16 + n] = (__bf16)s1[r];
    }
    asm volatile("s_wait_dscnt 0" ::: "memory");
    v16bf ap;
    {
      const __bf16* pr = &pbuf[w][n][kA];
      ap = cat16(*(const v8bf*)pr, *(const v8bf*)(pr + 8));
    }
#pragma unroll
    for (int nt = 0; nt < 4; ++nt)
      oacc[nt] = wmma_bf16(ap, vb[nt], oacc[nt]);
  }

  // normalize and store bf16 into (B, T, H*DK) for final projection
#pragma unroll
  for (int r = 0; r < 8; ++r) {
    const float inv = 1.0f / lrow[r];
    const int t = t0 + r + mo;
#pragma unroll
    for (int nt = 0; nt < 4; ++nt) {
      const int col = h * DK + nt * 16 + n;
      Obuf[((size_t)(b * TSEQ + t)) * HID + col] = (__bf16)(oacc[nt][r] * inv);
    }
  }
}

// ---------------- host launch ----------------

extern "C" void kernel_launch(void* const* d_in, const int* in_sizes, int n_in,
                              void* d_out, int out_size, void* d_ws, size_t ws_size,
                              hipStream_t stream) {
  (void)in_sizes; (void)n_in; (void)out_size; (void)ws_size;
  const float* x   = (const float*)d_in[0];
  const float* pos = (const float*)d_in[1];
  const float* Wq  = (const float*)d_in[2];
  const float* bq  = (const float*)d_in[3];
  const float* Wk  = (const float*)d_in[4];
  const float* bk  = (const float*)d_in[5];
  const float* Wv  = (const float*)d_in[6];
  const float* bv  = (const float*)d_in[7];
  const float* Wp  = (const float*)d_in[8];
  const float* bp  = (const float*)d_in[9];
  const float* Wo  = (const float*)d_in[10];
  const float* bo  = (const float*)d_in[11];
  const float* pbu = (const float*)d_in[12];
  const float* pbv = (const float*)d_in[13];
  float* out = (float*)d_out;

  char* ws = (char*)d_ws;
  size_t off = 0;
  auto take = [&](size_t bytes) -> char* {
    char* p = ws + off;
    off = (off + bytes + 255) & ~(size_t)255;
    return p;
  };
  const size_t NTOK = (size_t)NB * TSEQ;            // 4096
  __bf16* xbf   = (__bf16*)take(NTOK * HID * 2);
  __bf16* posbf = (__bf16*)take((size_t)PPAD * HID * 2);
  __bf16* WqT   = (__bf16*)take((size_t)HID * HID * 2);
  __bf16* WkT   = (__bf16*)take((size_t)HID * HID * 2);
  __bf16* WvT   = (__bf16*)take((size_t)HID * HID * 2);
  __bf16* WpT   = (__bf16*)take((size_t)HID * HID * 2);
  __bf16* WoT   = (__bf16*)take((size_t)HID * HID * 2);
  float*  qf    = (float*)take(NTOK * HID * 4);
  float*  kf    = (float*)take(NTOK * HID * 4);
  float*  vf    = (float*)take(NTOK * HID * 4);
  float*  pf    = (float*)take((size_t)PPAD * HID * 4);
  __bf16* Qu    = (__bf16*)take(NTOK * HID * 2);
  __bf16* Qv    = (__bf16*)take(NTOK * HID * 2);
  __bf16* Kh    = (__bf16*)take(NTOK * HID * 2);
  __bf16* Vt    = (__bf16*)take(NTOK * HID * 2);
  __bf16* Ph    = (__bf16*)take((size_t)NH * PPAD * DK * 2);
  __bf16* Obuf  = (__bf16*)take(NTOK * HID * 2);

  const int TPB = 256;
  const int nX = (int)(NTOK * HID);                 // 2,097,152
  const int nP = PPOS * HID;                        // 1,048,064

  cvt_bf16_kernel<<<(nX + TPB - 1) / TPB, TPB, 0, stream>>>(x, xbf, nX);
  cvt_bf16_kernel<<<(nP + TPB - 1) / TPB, TPB, 0, stream>>>(pos, posbf, nP);
  transpose_w_kernel<<<HID * HID / TPB, TPB, 0, stream>>>(Wq, WqT);
  transpose_w_kernel<<<HID * HID / TPB, TPB, 0, stream>>>(Wk, WkT);
  transpose_w_kernel<<<HID * HID / TPB, TPB, 0, stream>>>(Wv, WvT);
  transpose_w_kernel<<<HID * HID / TPB, TPB, 0, stream>>>(Wp, WpT);
  transpose_w_kernel<<<HID * HID / TPB, TPB, 0, stream>>>(Wo, WoT);

  gemm_bf16_kernel<<<(int)(NTOK / 64), TPB, 0, stream>>>(xbf, WqT, bq, qf);
  gemm_bf16_kernel<<<(int)(NTOK / 64), TPB, 0, stream>>>(xbf, WkT, bk, kf);
  gemm_bf16_kernel<<<(int)(NTOK / 64), TPB, 0, stream>>>(xbf, WvT, bv, vf);
  gemm_bf16_kernel<<<PPAD / 64, TPB, 0, stream>>>(posbf, WpT, bp, pf);

  scatter_q_kernel<<<nX / TPB, TPB, 0, stream>>>(qf, pbu, pbv, Qu, Qv);
  scatter_kv_kernel<<<nX / TPB, TPB, 0, stream>>>(kf, vf, Kh, Vt);
  scatter_p_kernel<<<(nP + TPB - 1) / TPB, TPB, 0, stream>>>(pf, Ph);

  attn_kernel<<<NB * NH * (TSEQ / 16) / 8, TPB, 0, stream>>>(Qu, Qv, Kh, Vt, Ph, Obuf);

  gemm_bf16_kernel<<<(int)(NTOK / 64), TPB, 0, stream>>>(Obuf, WoT, bo, out);
}
